// MultiHeadAttention_47450798686789
// MI455X (gfx1250) — matrix operand order repaired and checked by reading, not compiled, not run on hardware
//
#include <hip/hip_runtime.h>
#include <hip/hip_bf16.h>

// Problem constants (fixed by the reference)
#define BB 2
#define SS 2048
#define DD 1024
#define HH 16
#define HDIM 64

typedef __attribute__((ext_vector_type(16))) __bf16 v16bf;
typedef __attribute__((ext_vector_type(8)))  __bf16 bf16x8;
typedef __attribute__((ext_vector_type(4)))  __bf16 bf16x4;
typedef __attribute__((ext_vector_type(8)))  float  v8f;
typedef __attribute__((ext_vector_type(4)))  unsigned int u32x4;
typedef __attribute__((ext_vector_type(8)))  int i32x8;
typedef __attribute__((ext_vector_type(4)))  int i32x4;

// Tensor Data Mover availability (5-arg builtin confirmed working this round:
// asm shows "tensor_load_to_lds s[..],s[..]" + "s_wait_tensorcnt 0x0").
#if defined(__has_builtin)
#  if __has_builtin(__builtin_amdgcn_tensor_load_to_lds) && \
      __has_builtin(__builtin_amdgcn_s_wait_tensorcnt)
#    define USE_TDM 1
#  endif
#endif
#ifndef USE_TDM
#  define USE_TDM 0
#endif

static __device__ __forceinline__ v8f wmma_bf16(v16bf a, v16bf b, v8f c) {
    // D = A(16x32 bf16) * B(32x16 bf16) + C(16x16 f32)
    return __builtin_amdgcn_wmma_f32_16x16x32_bf16(
        /*neg_a=*/false, a, /*neg_b=*/false, b,
        /*c_mod=*/(short)0, c, /*reuse_a=*/false, /*reuse_b=*/false);
}

static __device__ __forceinline__ v16bf cat8(bf16x8 lo, bf16x8 hi) {
    return __builtin_shufflevector(lo, hi, 0,1,2,3,4,5,6,7,8,9,10,11,12,13,14,15);
}

static __device__ __forceinline__ bf16x8 ld8(const __bf16* p) {
    return *(const bf16x8*)p;   // 16B load (global_load_b128 / ds_load_b128)
}

// ---------------------------------------------------------------------------
// fp32 -> bf16 conversion (n is a multiple of 4*256 for all arrays here)
// ---------------------------------------------------------------------------
__global__ __launch_bounds__(256)
void cvt_f32_bf16(const float* __restrict__ src, __bf16* __restrict__ dst) {
    size_t i = ((size_t)blockIdx.x * 256 + threadIdx.x) * 4;
    float4 f = *(const float4*)(src + i);
    bf16x4 o;
    o[0] = (__bf16)f.x; o[1] = (__bf16)f.y; o[2] = (__bf16)f.z; o[3] = (__bf16)f.w;
    *(bf16x4*)(dst + i) = o;
}

// ---------------------------------------------------------------------------
// NT GEMM:  C[m,n] = sum_k A[m,k] * W[n,k]     (A: [M,1024], W: [1024,1024])
// Block tile 128x128, K-step 64 (two 16x16x32 chunks per stage), 8 waves with
// 2x4 WMMA tiles each -> 16 WMMAs per barrier. Tile staging via Tensor Data
// Mover (double-buffered, TENSORcnt) when available; VGPR staging fallback.
// Epilogue modes:
//   0: fp32, row-major [M, D]                 (final projection -> d_out)
//   1: bf16, scatter to [B,H,S,HD]            (K projection)
//   2: bf16, scatter to [B,H,HD,S] transposed (V projection)
//   3: mode 1 with *0.125 softmax scale baked (Q projection)
// ---------------------------------------------------------------------------
#define BM 128
#define BN 128
#define BKK 64
#define LDSP 72   // padded LDS row stride: 64 elems + 8 pad -> 144B, 16B-aligned
#define ALDSP 40  // attention P-staging stride: 32 elems + 8 pad -> 80B

#if USE_TDM
// Loop-invariant part of the tensor descriptor (D# group 1), ISA ch.8 §8.4.
static __device__ __forceinline__ i32x8 tdm_group1(int tensor_rows) {
    i32x8 g1;
    g1[0] = (1 << 16)       // data_size = 2 bytes
          | (1 << 20)       // pad_enable (loads only)
          | (4 << 22)       // pad_interval: 32 DWORDs (=128B = one tile row)
          | (3 << 25);      // pad_amount:   4 DWORDs (=16B -> stride 72 elems)
    g1[1] = (DD & 0xFFFF) << 16;                          // tensor_dim0 lo
    g1[2] = (DD >> 16) | ((tensor_rows & 0xFFFF) << 16);  // dim0 hi | dim1 lo
    g1[3] = ((unsigned)tensor_rows >> 16) | (BKK << 16);  // dim1 hi | tile_dim0=64
    g1[4] = BM;                                           // tile_dim1=128, dim2=0
    g1[5] = DD;                                           // tensor_dim0_stride lo
    g1[6] = 0;                                            // stride0 hi | stride1 lo
    g1[7] = 0;
    return g1;
}

// Issue one TDM 2D tile load (128 rows x 64 bf16) to LDS with row padding.
static __device__ __forceinline__ void tdm_issue(const __bf16* gsrc,
                                                 unsigned lds_byte_off,
                                                 i32x8 g1) {
    unsigned long long ga = (unsigned long long)(uintptr_t)gsrc;
    u32x4 g0;
    g0[0] = 1u;                                   // count=1 (valid user D#)
    g0[1] = lds_byte_off;                         // lds_addr [63:32]
    g0[2] = (unsigned)(ga & 0xFFFFFFFFu);         // global_addr low
    g0[3] = (unsigned)((ga >> 32) & 0x1FFFFFFu)   // global_addr [56:32]
          | (2u << 30);                           // type=2 ("image")
    i32x4 z4 = {};
#if __clang_major__ >= 23
    i32x8 z8 = {};
    __builtin_amdgcn_tensor_load_to_lds(g0, g1, z4, z4, z8, 0);
#else
    __builtin_amdgcn_tensor_load_to_lds(g0, g1, z4, z4, 0);
#endif
}
#endif  // USE_TDM

// One K=64 stage: load all fragments into distinct locals first (independent
// ds_load_b128s -> incremental s_wait_dscnt), then 8 WMMAs per K-chunk.
static __device__ __forceinline__ void mma_stage(const __bf16* __restrict__ Ab,
                                                 const __bf16* __restrict__ Bb,
                                                 int wrow, int wcol, int l15,
                                                 int hgrp, v8f acc[2][4]) {
    #pragma unroll
    for (int kk = 0; kk < 2; ++kk) {
        const int kb = kk * 32;
        const __bf16* ar0 = Ab + (wrow * 32 + l15) * LDSP + kb;
        const __bf16* ar1 = ar0 + 16 * LDSP;
        const __bf16* br  = Bb + (wcol * 64 + l15) * LDSP + kb + hgrp * 8;
        v16bf af0 = cat8(ld8(ar0 + hgrp * 8), ld8(ar0 + 16 + hgrp * 8));
        v16bf af1 = cat8(ld8(ar1 + hgrp * 8), ld8(ar1 + 16 + hgrp * 8));
        v16bf b0  = cat8(ld8(br),              ld8(br + 16));
        v16bf b1  = cat8(ld8(br + 16 * LDSP),  ld8(br + 16 * LDSP + 16));
        v16bf b2  = cat8(ld8(br + 32 * LDSP),  ld8(br + 32 * LDSP + 16));
        v16bf b3  = cat8(ld8(br + 48 * LDSP),  ld8(br + 48 * LDSP + 16));
        acc[0][0] = wmma_bf16(af0, b0, acc[0][0]);
        acc[1][0] = wmma_bf16(af1, b0, acc[1][0]);
        acc[0][1] = wmma_bf16(af0, b1, acc[0][1]);
        acc[1][1] = wmma_bf16(af1, b1, acc[1][1]);
        acc[0][2] = wmma_bf16(af0, b2, acc[0][2]);
        acc[1][2] = wmma_bf16(af1, b2, acc[1][2]);
        acc[0][3] = wmma_bf16(af0, b3, acc[0][3]);
        acc[1][3] = wmma_bf16(af1, b3, acc[1][3]);
    }
}

__global__ __launch_bounds__(256)
void gemm_nt(const __bf16* __restrict__ A, const __bf16* __restrict__ W,
             float* __restrict__ Cf, __bf16* __restrict__ Cb, int mode,
             int a_rows) {
    const int tid  = threadIdx.x;
    const int lane = tid & 31;
    const int wave = tid >> 5;
    const int l15  = lane & 15;
    const int hgrp = lane >> 4;          // 0 or 1 (16-lane half)
    const int m0   = blockIdx.y * BM;
    const int n0   = blockIdx.x * BN;
    const int wrow = wave >> 1;          // 0..3 : 32-row slab
    const int wcol = wave & 1;           // 0..1 : 64-col slab

    v8f zc = {};
    v8f acc[2][4];
    #pragma unroll
    for (int i = 0; i < 2; ++i)
        #pragma unroll
        for (int j = 0; j < 4; ++j) acc[i][j] = zc;

#if USE_TDM
    __shared__ __align__(16) __bf16 As[2][BM * LDSP];
    __shared__ __align__(16) __bf16 Bs[2][BN * LDSP];

    const unsigned aoff0 = (unsigned)(uintptr_t)(void*)&As[0][0];
    const unsigned aoff1 = (unsigned)(uintptr_t)(void*)&As[1][0];
    const unsigned boff0 = (unsigned)(uintptr_t)(void*)&Bs[0][0];
    const unsigned boff1 = (unsigned)(uintptr_t)(void*)&Bs[1][0];

    const i32x8 g1a = tdm_group1(a_rows);
    const i32x8 g1b = tdm_group1(DD);
    const __bf16* Abase = A + (size_t)m0 * DD;
    const __bf16* Wbase = W + (size_t)n0 * DD;

    // preload first K-tile via TDM (one wave issues; DMA engine does the rest)
    if (wave == 0) {
        tdm_issue(Abase, aoff0, g1a);
        tdm_issue(Wbase, boff0, g1b);
        __builtin_amdgcn_s_wait_tensorcnt(0);
    }
    __syncthreads();

    int cur = 0;
    for (int kt = 0; kt < DD; kt += BKK) {
        // kick off next tile's DMA while everyone computes this one
        if (wave == 0 && kt + BKK < DD) {
            tdm_issue(Abase + kt + BKK, cur ? aoff0 : aoff1, g1a);
            tdm_issue(Wbase + kt + BKK, cur ? boff0 : boff1, g1b);
        }
        mma_stage(&As[cur][0], &Bs[cur][0], wrow, wcol, l15, hgrp, acc);

        // next buffer ready + everyone done reading current buffer
        if (wave == 0) __builtin_amdgcn_s_wait_tensorcnt(0);
        __syncthreads();
        cur ^= 1;
    }
#else
    __shared__ __align__(16) __bf16 As[BM * LDSP];
    __shared__ __align__(16) __bf16 Bs[BN * LDSP];

    const int lr = tid >> 1;             // 0..127 tile row per thread
    const int lc = (tid & 1) * 32;       // 0 / 32 col chunk

    for (int kt = 0; kt < DD; kt += BKK) {
        {
            const __bf16* ga = A + (size_t)(m0 + lr) * DD + kt + lc;
            bf16x8 a0 = ld8(ga), a1 = ld8(ga + 8), a2 = ld8(ga + 16), a3 = ld8(ga + 24);
            *(bf16x8*)(As + lr * LDSP + lc)      = a0;
            *(bf16x8*)(As + lr * LDSP + lc + 8)  = a1;
            *(bf16x8*)(As + lr * LDSP + lc + 16) = a2;
            *(bf16x8*)(As + lr * LDSP + lc + 24) = a3;
            const __bf16* gb = W + (size_t)(n0 + lr) * DD + kt + lc;
            bf16x8 b0 = ld8(gb), b1 = ld8(gb + 8), b2 = ld8(gb + 16), b3 = ld8(gb + 24);
            *(bf16x8*)(Bs + lr * LDSP + lc)      = b0;
            *(bf16x8*)(Bs + lr * LDSP + lc + 8)  = b1;
            *(bf16x8*)(Bs + lr * LDSP + lc + 16) = b2;
            *(bf16x8*)(Bs + lr * LDSP + lc + 24) = b3;
        }
        __syncthreads();
        mma_stage(As, Bs, wrow, wcol, l15, hgrp, acc);
        __syncthreads();
    }
#endif

    // epilogue: C layout row = (hgrp ? v+8 : v), col = l15
    #pragma unroll
    for (int tr = 0; tr < 2; ++tr)
        #pragma unroll
        for (int tc = 0; tc < 4; ++tc) {
            const int rbase = m0 + wrow * 32 + tr * 16;
            const int cbase = n0 + wcol * 64 + tc * 16;
            #pragma unroll
            for (int v = 0; v < 8; ++v) {
                int row = rbase + (hgrp ? v + 8 : v);
                int col = cbase + l15;
                float val = acc[tr][tc][v];
                if (mode == 0) {
                    Cf[(size_t)row * DD + col] = val;
                } else {
                    int b = row >> 11, s = row & (SS - 1);
                    int h = col >> 6,  hd = col & (HDIM - 1);
                    if (mode == 3) val *= 0.125f;   // 1/sqrt(64) baked into Q
                    size_t idx = (mode == 2)
                        ? ((size_t)((b * HH + h) * HDIM + hd)) * SS + s     // V^T
                        : ((size_t)((b * HH + h) * SS + s)) * HDIM + hd;    // Q,K
                    Cb[idx] = (__bf16)val;
                }
            }
        }
}

// ---------------------------------------------------------------------------
// Flash attention, causal. 128 threads = 4 independent waves; each wave owns
// 16 query rows and streams keys in chunks of 32 (no block barriers: per-wave
// causal trip counts). Q scale pre-baked. Output bf16 [B,S,D].
// ---------------------------------------------------------------------------
__global__ __launch_bounds__(128)
void attn(const __bf16* __restrict__ Qh, const __bf16* __restrict__ Kh,
          const __bf16* __restrict__ Vt, __bf16* __restrict__ O) {
    __shared__ __align__(16) __bf16 pbuf[4][16][ALDSP];  // per-wave P staging

    const int lane  = threadIdx.x & 31;
    const int wave  = threadIdx.x >> 5;
    const int l15   = lane & 15;
    const int hgrp  = lane >> 4;
    const int qbase = blockIdx.x * 64 + wave * 16;
    const int h     = blockIdx.y;
    const int b     = blockIdx.z;

    const __bf16* qp = Qh + (size_t)(b * HH + h) * SS * HDIM;
    const __bf16* kp = Kh + (size_t)(b * HH + h) * SS * HDIM;
    const __bf16* vp = Vt + (size_t)(b * HH + h) * HDIM * SS;

    // Q A-fragments (loop invariant): frag f covers hd = f*32 .. f*32+31
    v16bf aq[2];
    {
        const __bf16* qr = qp + (size_t)(qbase + l15) * HDIM;
        #pragma unroll
        for (int f = 0; f < 2; ++f)
            aq[f] = cat8(ld8(qr + f * 32 + hgrp * 8),
                         ld8(qr + f * 32 + 16 + hgrp * 8));
    }

    float mst[8], lst[8];
    v8f zc = {};
    v8f accO[4];
    #pragma unroll
    for (int j = 0; j < 4; ++j) accO[j] = zc;
    #pragma unroll
    for (int v = 0; v < 8; ++v) { mst[v] = -1e30f; lst[v] = 0.0f; }

    const int qlast = qbase + 15;
    for (int kt = 0; kt <= qlast; kt += 32) {
        // prefetch next key chunk while this one computes (global_prefetch_b8)
        if (kt + 32 <= qlast) {
            __builtin_prefetch(kp + (size_t)(kt + 32 + lane) * HDIM, 0, 1);
            __builtin_prefetch(vp + (size_t)(lane * 2) * SS + kt + 32, 0, 1);
            __builtin_prefetch(vp + (size_t)(lane * 2 + 1) * SS + kt + 32, 0, 1);
        }

        // ---- scores: two 16x16 tiles (key groups of 16), contraction HD=64
        v8f s[2];
        #pragma unroll
        for (int ks = 0; ks < 2; ++ks) {
            const __bf16* kr = kp + (size_t)(kt + ks * 16 + l15) * HDIM + hgrp * 8;
            v16bf kb0 = cat8(ld8(kr),      ld8(kr + 16));       // hd 0..31
            v16bf kb1 = cat8(ld8(kr + 32), ld8(kr + 48));       // hd 32..63
            v8f z = zc;
            z = wmma_bf16(aq[0], kb0, z);
            z = wmma_bf16(aq[1], kb1, z);
            s[ks] = z;
        }

        // ---- online softmax in C-fragment layout
        float alpha[8];
        #pragma unroll
        for (int v = 0; v < 8; ++v) {
            const int qrow = qbase + (hgrp ? v + 8 : v);
            const int k0 = kt + l15, k1 = kt + 16 + l15;
            float s0 = (k0 <= qrow) ? s[0][v] : -1e30f;
            float s1 = (k1 <= qrow) ? s[1][v] : -1e30f;
            float rm = fmaxf(s0, s1);
            rm = fmaxf(rm, __shfl_xor(rm, 1, 32));
            rm = fmaxf(rm, __shfl_xor(rm, 2, 32));
            rm = fmaxf(rm, __shfl_xor(rm, 4, 32));
            rm = fmaxf(rm, __shfl_xor(rm, 8, 32));
            const float mnew = fmaxf(mst[v], rm);
            const float a  = __expf(mst[v] - mnew);
            const float p0 = __expf(s0 - mnew);
            const float p1 = __expf(s1 - mnew);
            float rs = p0 + p1;
            rs += __shfl_xor(rs, 1, 32);
            rs += __shfl_xor(rs, 2, 32);
            rs += __shfl_xor(rs, 4, 32);
            rs += __shfl_xor(rs, 8, 32);
            lst[v] = lst[v] * a + rs;
            mst[v] = mnew;
            alpha[v] = a;
            const int row = hgrp ? v + 8 : v;
            pbuf[wave][row][l15]      = (__bf16)p0;   // keys kt..kt+15
            pbuf[wave][row][16 + l15] = (__bf16)p1;   // keys kt+16..kt+31
        }
        // per-wave LDS store->load ordering (waves have different trip counts,
        // so no __syncthreads; CDNA5 split DS counter fence instead)
        asm volatile("s_wait_dscnt 0" ::: "memory");

        // rescale O accumulators (same row<->(hgrp,v) map as score tile)
        #pragma unroll
        for (int j = 0; j < 4; ++j)
            #pragma unroll
            for (int v = 0; v < 8; ++v) accO[j][v] *= alpha[v];

        // P as A-fragment (16 rows x 32 keys) from LDS
        const __bf16* pr = &pbuf[wave][l15][0];
        v16bf pa = cat8(ld8(pr + hgrp * 8), ld8(pr + 16 + hgrp * 8));

        // O += P @ V   (V transposed: [hd][s], key-contiguous per lane)
        #pragma unroll
        for (int j = 0; j < 4; ++j) {
            const __bf16* vr = vp + (size_t)(j * 16 + l15) * SS + kt + hgrp * 8;
            v16bf vb = cat8(ld8(vr), ld8(vr + 16));
            accO[j] = wmma_bf16(pa, vb, accO[j]);
        }
    }

    // normalize and store O as bf16 [B, S, D]
    #pragma unroll
    for (int j = 0; j < 4; ++j)
        #pragma unroll
        for (int v = 0; v < 8; ++v) {
            const int qrow = qbase + (hgrp ? v + 8 : v);
            const int col  = h * HDIM + j * 16 + l15;
            O[((size_t)(b * SS + qrow)) * DD + col] = (__bf16)(accO[j][v] / lst[v]);
        }
}

// ---------------------------------------------------------------------------
extern "C" void kernel_launch(void* const* d_in, const int* in_sizes, int n_in,
                              void* d_out, int out_size, void* d_ws, size_t ws_size,
                              hipStream_t stream) {
    const float* q_f  = (const float*)d_in[0];
    const float* k_f  = (const float*)d_in[1];
    const float* v_f  = (const float*)d_in[2];
    const float* wq_f = (const float*)d_in[3];
    const float* wk_f = (const float*)d_in[4];
    const float* wv_f = (const float*)d_in[5];
    const float* wo_f = (const float*)d_in[6];

    const size_t nX = (size_t)BB * SS * DD;   // 4,194,304
    const size_t nW = (size_t)DD * DD;        // 1,048,576

    __bf16* ws = (__bf16*)d_ws;               // 64 MB of bf16 scratch total
    __bf16* qx = ws;            __bf16* kx = qx + nX;  __bf16* vx = kx + nX;
    __bf16* wq = vx + nX;       __bf16* wk = wq + nW;
    __bf16* wv = wk + nW;       __bf16* wo = wv + nW;
    __bf16* Qh = wo + nW;       __bf16* Kh = Qh + nX;
    __bf16* Vt = Kh + nX;       __bf16* Ob = Vt + nX;

    const int gx = (int)(nX / 1024);          // 4096 blocks (4 elems/thread)
    const int gw = (int)(nW / 1024);          // 1024 blocks
    cvt_f32_bf16<<<gx, 256, 0, stream>>>(q_f,  qx);
    cvt_f32_bf16<<<gx, 256, 0, stream>>>(k_f,  kx);
    cvt_f32_bf16<<<gx, 256, 0, stream>>>(v_f,  vx);
    cvt_f32_bf16<<<gw, 256, 0, stream>>>(wq_f, wq);
    cvt_f32_bf16<<<gw, 256, 0, stream>>>(wk_f, wk);
    cvt_f32_bf16<<<gw, 256, 0, stream>>>(wv_f, wv);
    cvt_f32_bf16<<<gw, 256, 0, stream>>>(wo_f, wo);

    const int Mrows = BB * SS;                // 4096
    dim3 ggrid(DD / BN, Mrows / BM);          // (8, 32)
    gemm_nt<<<ggrid, 256, 0, stream>>>(qx, wq, nullptr, Qh, 3, Mrows); // Q
    gemm_nt<<<ggrid, 256, 0, stream>>>(kx, wk, nullptr, Kh, 1, Mrows); // K
    gemm_nt<<<ggrid, 256, 0, stream>>>(vx, wv, nullptr, Vt, 2, Mrows); // V^T

    dim3 agrid(SS / 64, HH, BB);              // (32, 16, 2)
    attn<<<agrid, 128, 0, stream>>>(Qh, Kh, Vt, Ob);

    gemm_nt<<<ggrid, 256, 0, stream>>>(Ob, wo, (float*)d_out, nullptr, 0, Mrows);
}
